// MultiHeadAttention_50757923504652
// MI455X (gfx1250) — compile-verified, matrix operand order checked by reading, not run on hardware
//
#include <hip/hip_runtime.h>
#include <hip/hip_bf16.h>
#include <math.h>

// ---------------------------------------------------------------------------
// MultiHeadAttention for MI455X (gfx1250): bf16 WMMA (16x16x32, fp32 accum)
// B=2, S=2048, D=1024, H=16, HD=64
// Round 5: fix async builtin signature — clang says param 1 is
// 'int __vector_size__(16) * (device AS)', non-const. Cast accordingly.
// Flash attention: cooperative LDS-staged K/V tiles filled with
// GLOBAL_LOAD_ASYNC_TO_LDS_B128 (ASYNCcnt-tracked), ds_load_b128 fragments.
// ---------------------------------------------------------------------------

#define BB 2
#define SS 2048
#define DD 1024
#define HH 16
#define HD 64
#define MTOT (BB * SS)       // 4096 rows of x
#define N_QKV (3 * DD)       // 3072
#define BH (BB * HH)         // 32

typedef __bf16 bf16_t;
typedef __attribute__((ext_vector_type(8)))  __bf16 v8bf;
typedef __attribute__((ext_vector_type(16))) __bf16 v16bf;
typedef __attribute__((ext_vector_type(8)))  float  v8f;
typedef __attribute__((__vector_size__(4 * sizeof(int)))) int v4i;

#if defined(__has_builtin)
#if __has_builtin(__builtin_amdgcn_global_load_async_to_lds_b128)
#define ATTN_ASYNC 1
#endif
#endif
#ifndef ATTN_ASYNC
#define ATTN_ASYNC 0
#endif

// 16-byte global -> LDS copy (async DMA when available)
__device__ __forceinline__ void copy16B_to_lds(bf16_t* dst, const bf16_t* src) {
#if ATTN_ASYNC
    __builtin_amdgcn_global_load_async_to_lds_b128(
        (__attribute__((address_space(1))) v4i*)src,
        (__attribute__((address_space(3))) v4i*)dst, 0, 0);
#else
    *(v8bf*)dst = *(const v8bf*)src;
#endif
}

// wait until all but the most recent 8 async copies of this wave are done
__device__ __forceinline__ void wait_async_prev8() {
#if ATTN_ASYNC
#if defined(__has_builtin) && __has_builtin(__builtin_amdgcn_s_wait_asynccnt)
    __builtin_amdgcn_s_wait_asynccnt(8);
#else
    asm volatile("s_wait_asynccnt 0x8" ::: "memory");
#endif
#endif
}
__device__ __forceinline__ void wait_async_all() {
#if ATTN_ASYNC
#if defined(__has_builtin) && __has_builtin(__builtin_amdgcn_s_wait_asynccnt)
    __builtin_amdgcn_s_wait_asynccnt(0);
#else
    asm volatile("s_wait_asynccnt 0x0" ::: "memory");
#endif
#endif
}

// fp32 -> bf16 round-to-nearest-even
__device__ __forceinline__ bf16_t f2bf(float f) {
    union { float f; unsigned u; } uu; uu.f = f;
    unsigned r = uu.u + 0x7FFFu + ((uu.u >> 16) & 1u);
    unsigned short h = (unsigned short)(r >> 16);
    return __builtin_bit_cast(bf16_t, h);
}

__device__ __forceinline__ v16bf cat16(v8bf lo, v8bf hi) {
    return __builtin_shufflevector(lo, hi, 0,1,2,3,4,5,6,7,8,9,10,11,12,13,14,15);
}

// A/B WMMA fragment (16x32 bf16) from a row-major row pointer.
// Lanes 0-15 take K chunks {0..7,16..23}, lanes 16-31 take {8..15,24..31}.
__device__ __forceinline__ v16bf load_frag(const bf16_t* row, int off0, int off1) {
    v8bf lo = *(const v8bf*)(row + off0);
    v8bf hi = *(const v8bf*)(row + off1);
    return cat16(lo, hi);
}

__device__ __forceinline__ v8f wmma_bf16(v16bf a, v16bf b, v8f c) {
    return __builtin_amdgcn_wmma_f32_16x16x32_bf16(
        false, a, false, b, (short)0, c, false, false);
}

// Load 4 B fragments (N = n0..n0+63) for K chunk k0 from row-major W(N,K)
__device__ __forceinline__ void load_btile(const bf16_t* wb, int n0, int lm,
                                           int sel, int k0, v16bf (&b)[4]) {
    #pragma unroll
    for (int nt = 0; nt < 4; ++nt) {
        const bf16_t* brow = wb + (size_t)(n0 + nt * 16 + lm) * DD;
        b[nt] = load_frag(brow, k0 + sel * 8, k0 + 16 + sel * 8);
    }
}

__device__ __forceinline__ void load_atile(const bf16_t* a0row, const bf16_t* a1row,
                                           int sel, int k0, v16bf& a0, v16bf& a1) {
    a0 = load_frag(a0row, k0 + sel * 8, k0 + 16 + sel * 8);
    a1 = load_frag(a1row, k0 + sel * 8, k0 + 16 + sel * 8);
}

__device__ __forceinline__ void gemm_step(const v16bf& a0, const v16bf& a1,
                                          const v16bf (&b)[4], v8f (&acc)[8]) {
    #pragma unroll
    for (int nt = 0; nt < 4; ++nt) {
        acc[nt]     = wmma_bf16(a0, b[nt], acc[nt]);
        acc[4 + nt] = wmma_bf16(a1, b[nt], acc[4 + nt]);
    }
}

// Software-pipelined 32(M) x 64(N) x 1024(K) wave GEMM, alternating buffers
__device__ __forceinline__ void gemm_32x64(
    const bf16_t* a0row, const bf16_t* a1row, const bf16_t* wb,
    int n0, int lm, int sel, v8f (&acc)[8])
{
    v16bf a0A, a1A, bA[4], a0B, a1B, bB[4];
    load_atile(a0row, a1row, sel, 0, a0A, a1A);
    load_btile(wb, n0, lm, sel, 0, bA);

    for (int k0 = 0; k0 < DD - 64; k0 += 64) {
        load_atile(a0row, a1row, sel, k0 + 32, a0B, a1B);
        load_btile(wb, n0, lm, sel, k0 + 32, bB);
        gemm_step(a0A, a1A, bA, acc);

        load_atile(a0row, a1row, sel, k0 + 64, a0A, a1A);
        load_btile(wb, n0, lm, sel, k0 + 64, bA);
        gemm_step(a0B, a1B, bB, acc);
    }
    load_atile(a0row, a1row, sel, DD - 32, a0B, a1B);
    load_btile(wb, n0, lm, sel, DD - 32, bB);
    gemm_step(a0A, a1A, bA, acc);
    gemm_step(a0B, a1B, bB, acc);
}

// ---------------------------------------------------------------------------
// Stage 0: fp32 -> bf16 conversion
// ---------------------------------------------------------------------------
__global__ __launch_bounds__(256) void cvt_f32_to_bf16(
    const float* __restrict__ src, bf16_t* __restrict__ dst, int n)
{
    int i = blockIdx.x * blockDim.x + threadIdx.x;
    if (i < n) dst[i] = f2bf(src[i]);
}

// ---------------------------------------------------------------------------
// Stage 1: QKV projection. qkv[m,n] = sum_k x[m,k]*Wqkv[n,k] + bqkv[n]
// ---------------------------------------------------------------------------
__global__ __launch_bounds__(256) void qkv_gemm(
    const bf16_t* __restrict__ xb,    // (4096,1024)
    const bf16_t* __restrict__ wb,    // (3072,1024)
    const float*  __restrict__ bias,  // (3072)
    bf16_t* __restrict__ qb,          // (BH, S, HD)
    bf16_t* __restrict__ kb,          // (BH, S, HD)
    bf16_t* __restrict__ vtb)         // (BH, HD, S)
{
    const int wave = threadIdx.x >> 5;
    const int lane = threadIdx.x & 31;
    const int lm   = lane & 15;
    const int sel  = lane >> 4;

    const int strips_n = N_QKV / 64;  // 48
    const int strip = blockIdx.x * 8 + wave;
    const int m0 = (strip / strips_n) * 32;
    const int n0 = (strip % strips_n) * 64;

    v8f acc[8] = {v8f{}, v8f{}, v8f{}, v8f{}, v8f{}, v8f{}, v8f{}, v8f{}};
    gemm_32x64(xb + (size_t)(m0 + lm) * DD,
               xb + (size_t)(m0 + 16 + lm) * DD,
               wb, n0, lm, sel, acc);

    #pragma unroll
    for (int nt = 0; nt < 4; ++nt) {
        const int n   = n0 + nt * 16 + lm;
        const float bv = bias[n];
        const int h   = n / 192;
        const int r3  = n % 192;
        const int sub = r3 >> 6;     // 0=Q, 1=K, 2=V (uniform over 16-wide tile)
        const int d   = r3 & 63;
        #pragma unroll
        for (int mt = 0; mt < 2; ++mt) {
            #pragma unroll
            for (int r = 0; r < 8; ++r) {
                const int gm = m0 + mt * 16 + r + 8 * sel;
                const int b_ = gm >> 11;
                const int s_ = gm & (SS - 1);
                const bf16_t v = f2bf(acc[mt * 4 + nt][r] + bv);
                const int bh = b_ * HH + h;
                if (sub == 0)      qb [((size_t)bh * SS + s_) * HD + d] = v;
                else if (sub == 1) kb [((size_t)bh * SS + s_) * HD + d] = v;
                else               vtb[((size_t)bh * HD + d) * SS + s_] = v;
            }
        }
    }
}

// ---------------------------------------------------------------------------
// Stage 2: flash attention, cooperative LDS-staged.
// Block = 4 waves, all on the same (b,h) (blockIdx>>4), each wave owns a
// 32-query tile. Double-buffered 64-key K/V^T tiles in LDS, filled by the
// whole block with async global->LDS copies (8 per wave per fill).
// ---------------------------------------------------------------------------
__global__ __launch_bounds__(128) void flash_attn(
    const bf16_t* __restrict__ qb,
    const bf16_t* __restrict__ kb,
    const bf16_t* __restrict__ vtb,
    const float*  __restrict__ mask,   // (S,S) broadcast
    bf16_t* __restrict__ valb)         // (B, S, D) = (b, s, h*64+d)
{
    __shared__ bf16_t Kt[2][64][64];   // 16 KB: local key row x hd
    __shared__ bf16_t Vt[2][64][64];   // 16 KB: hd row x local key col
    __shared__ bf16_t P[4][32][32];    //  8 KB: per-wave prob tiles

    const int tid  = threadIdx.x;
    const int wave = tid >> 5;
    const int lane = tid & 31;
    const int lm   = lane & 15;
    const int sel  = lane >> 4;

    const int bh = blockIdx.x >> 4;               // uniform over the block
    const int q0 = ((blockIdx.x * 4 + wave) & 63) * 32;

    // cooperative fill of one 64-key buffer (K rows + V^T rows), 8 asyncs/wave
    auto fill = [&](int buf, int j) {
        #pragma unroll
        for (int c = tid; c < 512; c += 128) {     // 64 rows x 8 chunks
            const int row  = c >> 3;
            const int col8 = (c & 7) * 8;
            copy16B_to_lds(&Kt[buf][row][col8],
                           kb + ((size_t)bh * SS + j + row) * HD + col8);
        }
        #pragma unroll
        for (int c = tid; c < 512; c += 128) {
            const int row  = c >> 3;
            const int col8 = (c & 7) * 8;
            copy16B_to_lds(&Vt[buf][row][col8],
                           vtb + ((size_t)bh * HD + row) * SS + j + col8);
        }
    };

    // Q fragments: [qtile][kstep], loop-invariant
    v16bf aq[2][2];
    #pragma unroll
    for (int qt = 0; qt < 2; ++qt) {
        const bf16_t* qrow = qb + ((size_t)bh * SS + q0 + 16 * qt + lm) * HD;
        aq[qt][0] = load_frag(qrow,      sel * 8, 16 + sel * 8);
        aq[qt][1] = load_frag(qrow, 32 + sel * 8, 48 + sel * 8);
    }

    float mi[2][8], li[2][8];
    v8f o[2][4];
    #pragma unroll
    for (int qt = 0; qt < 2; ++qt)
        #pragma unroll
        for (int r = 0; r < 8; ++r) { mi[qt][r] = -INFINITY; li[qt][r] = 0.0f; }
    #pragma unroll
    for (int qt = 0; qt < 2; ++qt)
        #pragma unroll
        for (int nt = 0; nt < 4; ++nt) o[qt][nt] = v8f{};

    const float scale = 0.125f;  // 1/sqrt(64)

    // scores + online softmax + PV for 32 keys at LDS half `half` (global j)
    auto process32 = [&](int buf, int half, int j) {
        v8f c[2][2];
        #pragma unroll
        for (int qt = 0; qt < 2; ++qt) {
            #pragma unroll
            for (int kt = 0; kt < 2; ++kt) {
                const bf16_t* krow = &Kt[buf][half * 32 + 16 * kt + lm][0];
                v8f cc = v8f{};
                cc = wmma_bf16(aq[qt][0],
                               load_frag(krow,      sel * 8, 16 + sel * 8), cc);
                cc = wmma_bf16(aq[qt][1],
                               load_frag(krow, 32 + sel * 8, 48 + sel * 8), cc);
                c[qt][kt] = cc;
            }
        }

        #pragma unroll
        for (int qt = 0; qt < 2; ++qt) {
            #pragma unroll
            for (int r = 0; r < 8; ++r) {
                const int qg = q0 + qt * 16 + r + 8 * sel;
                const float s0 = c[qt][0][r] * scale + mask[(size_t)qg * SS + j + lm];
                const float s1 = c[qt][1][r] * scale + mask[(size_t)qg * SS + j + 16 + lm];

                float mx = fmaxf(s0, s1);
                mx = fmaxf(mx, __shfl_xor(mx, 1));
                mx = fmaxf(mx, __shfl_xor(mx, 2));
                mx = fmaxf(mx, __shfl_xor(mx, 4));
                mx = fmaxf(mx, __shfl_xor(mx, 8));

                const float mnew  = fmaxf(mi[qt][r], mx);
                const float alpha = __expf(mi[qt][r] - mnew);
                const float p0 = __expf(s0 - mnew);
                const float p1 = __expf(s1 - mnew);

                float rs = p0 + p1;
                rs += __shfl_xor(rs, 1);
                rs += __shfl_xor(rs, 2);
                rs += __shfl_xor(rs, 4);
                rs += __shfl_xor(rs, 8);

                li[qt][r] = li[qt][r] * alpha + rs;
                mi[qt][r] = mnew;
                #pragma unroll
                for (int nt = 0; nt < 4; ++nt) o[qt][nt][r] *= alpha;

                P[wave][qt * 16 + r + 8 * sel][lm]      = f2bf(p0);
                P[wave][qt * 16 + r + 8 * sel][lm + 16] = f2bf(p1);
            }
        }

        #pragma unroll
        for (int qt = 0; qt < 2; ++qt) {
            const v16bf pa = load_frag(&P[wave][qt * 16 + lm][0],
                                       sel * 8, 16 + sel * 8);
            #pragma unroll
            for (int nt = 0; nt < 4; ++nt) {
                const bf16_t* vrow = &Vt[buf][nt * 16 + lm][half * 32];
                o[qt][nt] = wmma_bf16(pa,
                                      load_frag(vrow, sel * 8, 16 + sel * 8),
                                      o[qt][nt]);
            }
        }
    };

    // double-buffered 64-key loop
    fill(0, 0);
    for (int j0 = 0; j0 < SS; j0 += 64) {
        const int cur = (j0 >> 6) & 1;
        const int nxt = cur ^ 1;
        if (j0 + 64 < SS) {
            fill(nxt, j0 + 64);       // 8 new asyncs in flight
            wait_async_prev8();       // previous fill (cur) complete
        } else {
            wait_async_all();
        }
        __syncthreads();              // cur visible to all waves

        process32(cur, 0, j0);
        process32(cur, 1, j0 + 32);

        __syncthreads();              // all waves done reading cur
    }

    // ---- finalize: O /= l, write bf16 values (b, s, h*64+d)
    const int b_ = bh >> 4;
    const int h  = bh & 15;
    #pragma unroll
    for (int qt = 0; qt < 2; ++qt) {
        #pragma unroll
        for (int r = 0; r < 8; ++r) {
            const float inv = 1.0f / li[qt][r];
            const int s_ = q0 + qt * 16 + r + 8 * sel;
            #pragma unroll
            for (int nt = 0; nt < 4; ++nt) {
                valb[((size_t)(b_ * SS + s_)) * DD + h * HD + nt * 16 + lm] =
                    f2bf(o[qt][nt][r] * inv);
            }
        }
    }
}

// ---------------------------------------------------------------------------
// Stage 3: output projection. out[m,n] = sum_k val[m,k]*Wo[n,k] + bo[n]
// ---------------------------------------------------------------------------
__global__ __launch_bounds__(256) void out_gemm(
    const bf16_t* __restrict__ vb,   // (4096,1024)
    const bf16_t* __restrict__ wob,  // (1024,1024)
    const float*  __restrict__ bo,   // (1024)
    float* __restrict__ out)         // (4096,1024)
{
    const int wave = threadIdx.x >> 5;
    const int lane = threadIdx.x & 31;
    const int lm   = lane & 15;
    const int sel  = lane >> 4;

    const int strips_n = DD / 64;    // 16
    const int strip = blockIdx.x * 8 + wave;
    const int m0 = (strip / strips_n) * 32;
    const int n0 = (strip % strips_n) * 64;

    v8f acc[8] = {v8f{}, v8f{}, v8f{}, v8f{}, v8f{}, v8f{}, v8f{}, v8f{}};
    gemm_32x64(vb + (size_t)(m0 + lm) * DD,
               vb + (size_t)(m0 + 16 + lm) * DD,
               wob, n0, lm, sel, acc);

    #pragma unroll
    for (int nt = 0; nt < 4; ++nt) {
        const int n = n0 + nt * 16 + lm;
        const float bv = bo[n];
        #pragma unroll
        for (int mt = 0; mt < 2; ++mt) {
            #pragma unroll
            for (int r = 0; r < 8; ++r) {
                const int gm = m0 + mt * 16 + r + 8 * sel;
                out[(size_t)gm * DD + n] = acc[mt * 4 + nt][r] + bv;
            }
        }
    }
}

// ---------------------------------------------------------------------------
// Host launcher
// ---------------------------------------------------------------------------
extern "C" void kernel_launch(void* const* d_in, const int* in_sizes, int n_in,
                              void* d_out, int out_size, void* d_ws, size_t ws_size,
                              hipStream_t stream) {
    (void)in_sizes; (void)n_in; (void)out_size; (void)ws_size;

    const float* x    = (const float*)d_in[0];  // (B,S,D)
    const float* mask = (const float*)d_in[1];  // (1,1,S,S)
    const float* Wqkv = (const float*)d_in[2];  // (3D,D)
    const float* bqkv = (const float*)d_in[3];  // (3D)
    const float* Wo   = (const float*)d_in[4];  // (D,D)
    const float* bo   = (const float*)d_in[5];  // (D)
    float* out = (float*)d_out;

    char* p = (char*)d_ws;
    auto carve = [&p](size_t bytes) -> void* {
        void* r = (void*)p;
        p += (bytes + 255) & ~(size_t)255;
        return r;
    };
    const size_t nx   = (size_t)MTOT * DD;       // 4,194,304
    const size_t nwq  = (size_t)N_QKV * DD;      // 3,145,728
    const size_t nwo  = (size_t)DD * DD;         // 1,048,576
    const size_t nhd  = (size_t)BH * SS * HD;    // 4,194,304

    bf16_t* xb   = (bf16_t*)carve(nx  * 2);
    bf16_t* wqb  = (bf16_t*)carve(nwq * 2);
    bf16_t* wob  = (bf16_t*)carve(nwo * 2);
    bf16_t* qb   = (bf16_t*)carve(nhd * 2);
    bf16_t* kb   = (bf16_t*)carve(nhd * 2);
    bf16_t* vtb  = (bf16_t*)carve(nhd * 2);
    bf16_t* valb = (bf16_t*)carve(nx  * 2);

    cvt_f32_to_bf16<<<(int)((nx  + 255) / 256), 256, 0, stream>>>(x,    xb,  (int)nx);
    cvt_f32_to_bf16<<<(int)((nwq + 255) / 256), 256, 0, stream>>>(Wqkv, wqb, (int)nwq);
    cvt_f32_to_bf16<<<(int)((nwo + 255) / 256), 256, 0, stream>>>(Wo,   wob, (int)nwo);

    // Stage 1: (4096/32)*(3072/64) = 6144 waves, 8 waves/block
    qkv_gemm<<<6144 / 8, 256, 0, stream>>>(xb, wqb, bqkv, qb, kb, vtb);

    // Stage 2: 32 bh * 64 qtiles = 2048 waves, 4 waves/block (shared bh)
    flash_attn<<<2048 / 4, 128, 0, stream>>>(qb, kb, vtb, mask, valb);

    // Stage 3: (4096/32)*(1024/64) = 2048 waves, 8 waves/block
    out_gemm<<<2048 / 8, 256, 0, stream>>>(valb, wob, bo, out);
}
